// Memory_66133906424236
// MI455X (gfx1250) — matrix operand order repaired and checked by reading, not compiled, not run on hardware
//
#include <hip/hip_runtime.h>

typedef __attribute__((ext_vector_type(16))) __bf16 v16bf;
typedef __attribute__((ext_vector_type(8)))  float  v8f;

#define N_COMPS   68
#define BATCH     512
#define CCH       128
#define IMG_ELEMS (128 * 16 * 16)   // 32768 per image
#define NKCHUNK   36                // K = 9*128 = 1152 -> 36 chunks of 32
#define NCOTILE   8
#define PXSTRIDE  144               // bf16 elems per (ry,px) row in LDS (288B, 32B-aligned)

// ---------------------------------------------------------------------------
// Pack OIHW f32 weights into WMMA bf16 A-fragment order:
//   out[cotile][chunk][lane][j], chunk covers tap = chunk/4, ci = (chunk%4)*32 + kk
//   A 16x32 bf16 order used here (B matches): lane<16 row m holds K {0..15}; lane>=16 holds {16..31}
// ---------------------------------------------------------------------------
__global__ void pack_w(const float* __restrict__ w, __bf16* __restrict__ out) {
  int blk    = blockIdx.x;              // cotile*36 + chunk
  int cotile = blk / NKCHUNK;
  int chunk  = blk % NKCHUNK;
  int lane   = threadIdx.x;             // 0..31
  int m      = lane & 15;
  int kh     = lane >> 4;
  int tap    = chunk >> 2;
  int ci0    = (chunk & 3) * 32;
  int ky = tap / 3, kx = tap % 3;
  int co = cotile * 16 + m;
  __bf16* dst = out + ((size_t)blk * 32 + lane) * 16;
#pragma unroll
  for (int j = 0; j < 16; ++j) {
    int kk = j + kh * 16;   // within-chunk K index
    int ci = ci0 + kk;
    float v = w[(((size_t)co * CCH + ci) * 3 + ky) * 3 + kx];
    dst[j] = (__bf16)v;
  }
}

// ---------------------------------------------------------------------------
// 3x3 conv + bias + ReLU over 68 images (128ch, 16x16), implicit GEMM.
// grid = 68*16 (img, row y); block = 256 = 8 waves; wave = co-tile.
// LDS holds 3 padded input rows pre-converted to bf16; inner loop is pure
// ds_load_b128 + v_wmma (no per-chunk converts). Two accumulators break the
// WMMA dependency chain.
// ---------------------------------------------------------------------------
__global__ void __launch_bounds__(256)
conv3x3_relu(const float* __restrict__ in,      // [68][128][16][16]
             const __bf16* __restrict__ wpack,  // [8][36][32][16] bf16
             const float* __restrict__ bias,    // [128]
             float* __restrict__ out)           // [68][128][16][16]
{
  __shared__ __bf16 smem[3 * 18 * PXSTRIDE];    // [ry][px+pad][ci], bf16
  int img = blockIdx.x >> 4;
  int y   = blockIdx.x & 15;
  int tid = threadIdx.x;
  const float* src = in + (size_t)img * IMG_ELEMS;

  // Stage rows y-1..y+1 with x zero-padding into LDS, converting f32 -> bf16.
  for (int t = tid; t < 3 * 18 * CCH; t += 256) {
    int pxp = t % 18;
    int ci  = (t / 18) % CCH;
    int ry  = t / (18 * CCH);
    int row = y + ry - 1;
    int x   = pxp - 1;
    float v = 0.0f;
    if (row >= 0 && row < 16 && x >= 0 && x < 16)
      v = src[ci * 256 + row * 16 + x];
    smem[(ry * 18 + pxp) * PXSTRIDE + ci] = (__bf16)v;
  }
  __syncthreads();

  int wave = tid >> 5;                  // co-tile 0..7
  int lane = tid & 31;
  int px   = lane & 15;
  int kh   = lane >> 4;

  v8f acc0 = {};
  v8f acc1 = {};
  const __bf16* wbase = wpack + (size_t)wave * NKCHUNK * 32 * 16;
#pragma unroll 2
  for (int chunk = 0; chunk < NKCHUNK; chunk += 2) {
#pragma unroll
    for (int u = 0; u < 2; ++u) {
      int c   = chunk + u;
      v16bf afrag = *(const v16bf*)(wbase + ((size_t)c * 32 + lane) * 16);
      int tap = c >> 2;
      int ci0 = (c & 3) * 32 + kh * 16;   // B: lane<16 -> K 0..15, lane>=16 -> K 16..31
      int ry  = tap / 3;                  // dy + 1
      int xx  = px + (tap % 3);           // px + dx + 1 (padded coords)
      v16bf bfrag = *(const v16bf*)&smem[(ry * 18 + xx) * PXSTRIDE + ci0];
      if (u == 0)
        acc0 = __builtin_amdgcn_wmma_f32_16x16x32_bf16(false, afrag, false, bfrag,
                                                       (short)0, acc0, false, false);
      else
        acc1 = __builtin_amdgcn_wmma_f32_16x16x32_bf16(false, afrag, false, bfrag,
                                                       (short)0, acc1, false, false);
    }
  }

  // Epilogue: C/D layout -> co = wave*16 + r + (lane>=16 ? 8 : 0), pixel = lane&15.
  float* dst = out + (size_t)img * IMG_ELEMS;
#pragma unroll
  for (int r = 0; r < 8; ++r) {
    int co = wave * 16 + r + (kh ? 8 : 0);
    float v = acc0[r] + acc1[r] + bias[co];
    dst[co * 256 + y * 16 + px] = v > 0.0f ? v : 0.0f;
  }
}

// ---------------------------------------------------------------------------
// Per-output-item scatter-mean + gather of hypernet output. One block per (b,j).
// Deterministic, atomic-free: wave-0 ballot scan builds the ordered match list,
// then all threads stream float4s.
// ---------------------------------------------------------------------------
__global__ void __launch_bounds__(256)
scatter_mean_add(const int* __restrict__ style_ids,  // [512]
                 const int* __restrict__ comp_ids,   // [512][3]
                 const float* __restrict__ feats,    // [1536][32768]
                 const float* __restrict__ pb,       // [68][32768] conv hypernet out
                 float* __restrict__ outp)           // [1536][32768]
{
  __shared__ int skeys[BATCH * 3];
  __shared__ int slist[BATCH * 3];
  __shared__ int scount;
  int tid = threadIdx.x;

  for (int t = tid; t < BATCH * 3; t += 256) {
    int b = t / 3, j = t % 3;
    int off = (j == 0) ? 0 : ((j == 1) ? 19 : 40);
    skeys[t] = style_ids[b] * N_COMPS + comp_ids[t] + off;
  }
  __syncthreads();

  int o = blockIdx.x;
  if (tid < 32) {                        // wave-0 cooperative, deterministic scan
    int k = skeys[o];
    int c = 0;
    for (int base = 0; base < BATCH * 3; base += 32) {
      bool match = (skeys[base + tid] == k);
      unsigned long long msk = __ballot(match);
      if (tid == 0) {
        while (msk) {
          int b = __ffsll(msk) - 1;
          slist[c++] = base + b;
          msk &= msk - 1;
        }
      }
    }
    if (tid == 0) scount = c;
  }
  __syncthreads();

  int   cnt  = scount;                   // >= 1 always (item matches itself)
  float inv  = 1.0f / (float)cnt;
  int   addr = skeys[o] % N_COMPS;
  const float4* pbsrc4  = (const float4*)(pb + (size_t)addr * IMG_ELEMS);
  const float4* feats4  = (const float4*)feats;
  float4*       dst4    = (float4*)(outp + (size_t)o * IMG_ELEMS);

  for (int e = tid; e < IMG_ELEMS / 4; e += 256) {
    __builtin_prefetch(pbsrc4 + e + 256, 0, 1);
    float4 a = make_float4(0.f, 0.f, 0.f, 0.f);
    for (int m = 0; m < cnt; ++m) {
      const float4* fsrc = feats4 + (size_t)slist[m] * (IMG_ELEMS / 4);
      __builtin_prefetch(fsrc + e + 256, 0, 1);
      float4 t = fsrc[e];
      a.x += t.x; a.y += t.y; a.z += t.z; a.w += t.w;
    }
    float4 p = pbsrc4[e];
    float4 r;
    r.x = a.x * inv + p.x;
    r.y = a.y * inv + p.y;
    r.z = a.z * inv + p.z;
    r.w = a.w * inv + p.w;
    dst4[e] = r;
  }
}

// ---------------------------------------------------------------------------
extern "C" void kernel_launch(void* const* d_in, const int* in_sizes, int n_in,
                              void* d_out, int out_size, void* d_ws, size_t ws_size,
                              hipStream_t stream) {
  const int*   style_ids  = (const int*)d_in[0];
  const int*   comp_ids   = (const int*)d_in[1];
  const float* comp_feats = (const float*)d_in[2];
  const float* bias       = (const float*)d_in[3];
  const float* w1 = (const float*)d_in[4];
  const float* b1 = (const float*)d_in[5];
  const float* w2 = (const float*)d_in[6];
  const float* b2 = (const float*)d_in[7];
  const float* w3 = (const float*)d_in[8];
  const float* b3 = (const float*)d_in[9];

  char* ws = (char*)d_ws;
  const size_t PW = (size_t)NCOTILE * NKCHUNK * 32 * 16;   // 147456 bf16 per layer
  __bf16* pw1 = (__bf16*)ws;
  __bf16* pw2 = pw1 + PW;
  __bf16* pw3 = pw2 + PW;
  float* actA = (float*)(ws + 3 * PW * 2);                 // 68 images
  float* actB = actA + (size_t)N_COMPS * IMG_ELEMS;

  pack_w<<<NCOTILE * NKCHUNK, 32, 0, stream>>>(w1, pw1);
  pack_w<<<NCOTILE * NKCHUNK, 32, 0, stream>>>(w2, pw2);
  pack_w<<<NCOTILE * NKCHUNK, 32, 0, stream>>>(w3, pw3);

  conv3x3_relu<<<N_COMPS * 16, 256, 0, stream>>>(bias, pw1, b1, actA);
  conv3x3_relu<<<N_COMPS * 16, 256, 0, stream>>>(actA, pw2, b2, actB);
  conv3x3_relu<<<N_COMPS * 16, 256, 0, stream>>>(actB, pw3, b3, actA);

  scatter_mean_add<<<BATCH * 3, 256, 0, stream>>>(style_ids, comp_ids,
                                                  comp_feats, actA,
                                                  (float*)d_out);
}
